// GCL_64811056496980
// MI455X (gfx1250) — matrix operand order repaired and checked by reading, not compiled, not run on hardware
//
#include <hip/hip_runtime.h>

// ---------------------------------------------------------------------------
// bf16 WMMA GNN layer for gfx1250 (wave32), round 3:
//  - weights staged into LDS via the Tensor Data Mover (TENSORcnt), 6-arg
//    tensor_load_to_lds builtin form (clang-23 / therock-10.0 headers)
//  - each wave processes two 16-row M-tiles (32 edges/nodes) so every
//    B operand read from LDS feeds two v_wmma ops
//  - VMEM reserved for the random x-gathers, edge_feat streaming stores and
//    the fused segment-sum atomics (L2-resident agg)
// ---------------------------------------------------------------------------

typedef __bf16 bf16x;
typedef unsigned int u32;
typedef __attribute__((ext_vector_type(16))) __bf16 v16bf;
typedef __attribute__((ext_vector_type(8)))  __bf16 v8bf;
typedef __attribute__((ext_vector_type(8)))  float   v8f;
typedef __attribute__((ext_vector_type(4)))  u32     v4u;
typedef __attribute__((ext_vector_type(8)))  int     v8i;
typedef __attribute__((ext_vector_type(4)))  int     v4i;

#define LDS_STRIDE 72   // padded row stride (bf16 elems) for the transpose buf

#if __has_builtin(__builtin_amdgcn_tensor_load_to_lds)
#define USE_TDM 1
#else
#define USE_TDM 0
#endif

// ---------------- WMMA operand loaders -------------------------------------

// A operand (16x32 bf16, MxK): lane = M (0..15) per half-wave.
// elements 0..7  = K: kBase + 16*half       (VGPR0..3)
// elements 8..15 = K: kBase + 16*half + 8   (VGPR4..7)
static __device__ __forceinline__ v16bf load_a16(const bf16x* rowPtr, int kBase, int half) {
  v16bf a;
  v8bf* p = reinterpret_cast<v8bf*>(&a);
  p[0] = *reinterpret_cast<const v8bf*>(rowPtr + kBase + 16 * half);
  p[1] = *reinterpret_cast<const v8bf*>(rowPtr + kBase + 16 * half + 8);
  return a;
}

// B operand (32x16 bf16, KxN) from an N-major weight matrix (in LDS):
// lane = column n, 16 consecutive K values starting at kBase + 16*half.
static __device__ __forceinline__ v16bf load_b16(const bf16x* wt, int Ktot, int n,
                                                 int kBase, int half) {
  const bf16x* p0 = wt + n * Ktot + kBase + 16 * half;
  v16bf b;
  v8bf* p = reinterpret_cast<v8bf*>(&b);
  p[0] = *reinterpret_cast<const v8bf*>(p0);
  p[1] = *reinterpret_cast<const v8bf*>(p0 + 8);
  return b;
}

static __device__ __forceinline__ v8f wmma_bf16(v16bf a, v16bf b, v8f c) {
  return __builtin_amdgcn_wmma_f32_16x16x32_bf16(false, a, false, b, (short)0, c,
                                                 false, false);
}

// ---------------- Tensor Data Mover: 1-D bulk copy global->LDS -------------
// D# built per CDNA5 ISA (group0 + group1; groups 2/3 zero
// => 2D-or-less tensor).  elems = bf16 element count (tile_dim0, must fit 16b).

#if USE_TDM
static __device__ __forceinline__ void tdm_load_1d(u32 ldsOff, const void* gsrc, u32 elems) {
  unsigned long long ga = (unsigned long long)(size_t)gsrc;
  v4u g0;
  g0[0] = 1u;                                            // count=1, no gather
  g0[1] = ldsOff;                                        // lds_addr (bytes)
  g0[2] = (u32)ga;                                       // global_addr[31:0]
  g0[3] = (u32)((ga >> 32) & 0x01FFFFFFu) | (2u << 30);  // ga[56:32] | type=2
  v8i g1;
  g1[0] = (int)(1u << 16);                     // wg_mask=0, data_size=1 (2B)
  g1[1] = (int)((elems & 0xFFFFu) << 16);      // tensor_dim0[15:0]
  g1[2] = (int)(((elems >> 16) & 0xFFFFu) | (1u << 16));  // dim0 hi | tensor_dim1=1
  g1[3] = (int)((elems & 0xFFFFu) << 16);      // tile_dim0 = elems
  g1[4] = 0;                                   // tile_dim1=0 (1-D), tile_dim2=0
  g1[5] = (int)elems;                          // tensor_dim0_stride
  g1[6] = 0;
  g1[7] = 0;
  v4i z4 = {0, 0, 0, 0};
  v8i z8 = {0, 0, 0, 0, 0, 0, 0, 0};
  // 6-arg form: (group0, group1, group2, group3, group4(zero), cpol)
  __builtin_amdgcn_tensor_load_to_lds(g0, g1, z4, z4, z8, 0);
}
#endif

// Stage the two weight matrices of one MLP into LDS (whole block must call).
static __device__ __forceinline__ void stage_weights(bf16x* sW1, const bf16x* g1, int n1,
                                                     bf16x* sW2, const bf16x* g2, int n2) {
#if USE_TDM
  if ((threadIdx.x >> 5) == 0) {
    tdm_load_1d((u32)(size_t)(void*)sW1, g1, (u32)n1);
    tdm_load_1d((u32)(size_t)(void*)sW2, g2, (u32)n2);
    __builtin_amdgcn_s_wait_tensorcnt(0);
  }
#else
  for (int i = threadIdx.x; i < n1; i += blockDim.x) sW1[i] = g1[i];
  for (int i = threadIdx.x; i < n2; i += blockDim.x) sW2[i] = g2[i];
#endif
  __syncthreads();
}

// --------------------------- prep kernels ----------------------------------

__global__ void prep_x_agg(const float* __restrict__ x, bf16x* __restrict__ xb,
                           float* __restrict__ agg, int n) {
  int i = blockIdx.x * blockDim.x + threadIdx.x;
  if (i < n) {
    xb[i]  = (bf16x)x[i];
    agg[i] = 0.0f;
  }
}

__global__ void prep_w(const float* __restrict__ w, bf16x* __restrict__ wt,
                       int K, int N) {
  int i = blockIdx.x * blockDim.x + threadIdx.x;
  if (i < K * N) {
    int k = i / N;
    int n = i % N;
    wt[n * K + k] = (bf16x)w[i];   // row-major [K][N] -> N-major [N][K]
  }
}

__global__ void cvt_f2b(const float* __restrict__ s, bf16x* __restrict__ d, int n) {
  int i = blockIdx.x * blockDim.x + threadIdx.x;
  if (i < n) d[i] = (bf16x)s[i];
}

// --------------------------- edge kernel -----------------------------------
// Wave handles 32 edges (two 16-row M-tiles sharing every B operand).

__global__ void __launch_bounds__(256)
edge_kernel(const bf16x* __restrict__ xb, const int* __restrict__ row,
            const int* __restrict__ col,
            const bf16x* __restrict__ We1t, const float* __restrict__ be1,
            const bf16x* __restrict__ We2t, const float* __restrict__ be2,
            float* __restrict__ edge_out, float* __restrict__ agg, int E) {
  __shared__ bf16x sW1[128 * 64];                 // 16 KB, N-major [64][128]
  __shared__ bf16x sW2[64 * 64];                  //  8 KB, N-major [64][64]
  __shared__ bf16x sT[8 * 32 * LDS_STRIDE];       // 36 KB transpose buffers

  stage_weights(sW1, We1t, 128 * 64, sW2, We2t, 64 * 64);

  const int wave  = threadIdx.x >> 5;
  const int lane  = threadIdx.x & 31;
  const int laneM = lane & 15;
  const int half  = lane >> 4;

  const int tiles32 = (E + 31) >> 5;
  int tile = blockIdx.x * 8 + wave;
  const bool active = (tile < tiles32);
  if (!active) tile = 0;
  const int base = tile * 32;

  int eA0 = base + laneM;       if (eA0 > E - 1) eA0 = E - 1;
  int eA1 = base + 16 + laneM;  if (eA1 > E - 1) eA1 = E - 1;
  const bf16x* arow0 = xb + (size_t)row[eA0] * 64;
  const bf16x* acol0 = xb + (size_t)col[eA0] * 64;
  const bf16x* arow1 = xb + (size_t)row[eA1] * 64;
  const bf16x* acol1 = xb + (size_t)col[eA1] * 64;

  // ---- layer 1: [32 x 128] x [128 x 64] ----
  v8f acc[2][4] = {};
#pragma unroll
  for (int kt = 0; kt < 4; ++kt) {
    const int k = kt * 32;
    v16bf a0 = (k < 64) ? load_a16(arow0, k, half) : load_a16(acol0, k - 64, half);
    v16bf a1 = (k < 64) ? load_a16(arow1, k, half) : load_a16(acol1, k - 64, half);
#pragma unroll
    for (int t = 0; t < 4; ++t) {
      v16bf b = load_b16(sW1, 128, 16 * t + laneM, k, half);
      acc[0][t] = wmma_bf16(a0, b, acc[0][t]);
      acc[1][t] = wmma_bf16(a1, b, acc[1][t]);
    }
  }

  // bias + relu, transpose both M-tiles through LDS into A layout (bf16)
  bf16x* myT = sT + wave * 32 * LDS_STRIDE;
#pragma unroll
  for (int t = 0; t < 4; ++t) {
    const float bias = be1[16 * t + laneM];
#pragma unroll
    for (int mt = 0; mt < 2; ++mt) {
#pragma unroll
      for (int v = 0; v < 8; ++v) {
        float h = fmaxf(acc[mt][t][v] + bias, 0.0f);
        myT[(mt * 16 + v + 8 * half) * LDS_STRIDE + 16 * t + laneM] = (bf16x)h;
      }
    }
  }
  __syncthreads();

  // ---- layer 2: [32 x 64] x [64 x 64] ----
  v8f acc2[2][4] = {};
  const bf16x* at0 = myT + laneM * LDS_STRIDE;
  const bf16x* at1 = myT + (16 + laneM) * LDS_STRIDE;
#pragma unroll
  for (int kt = 0; kt < 2; ++kt) {
    const int k = kt * 32;
    v16bf a0 = load_a16(at0, k, half);
    v16bf a1 = load_a16(at1, k, half);
#pragma unroll
    for (int t = 0; t < 4; ++t) {
      v16bf b = load_b16(sW2, 64, 16 * t + laneM, k, half);
      acc2[0][t] = wmma_bf16(a0, b, acc2[0][t]);
      acc2[1][t] = wmma_bf16(a1, b, acc2[1][t]);
    }
  }

  // scatter indices: lane holds C rows M = mt*16 + v + 8*half
  int ridx[2][8];
#pragma unroll
  for (int mt = 0; mt < 2; ++mt)
#pragma unroll
    for (int v = 0; v < 8; ++v) {
      int e = base + mt * 16 + v + 8 * half;
      ridx[mt][v] = row[e > E - 1 ? E - 1 : e];
    }

#pragma unroll
  for (int t = 0; t < 4; ++t) {
    const float bias = be2[16 * t + laneM];
#pragma unroll
    for (int mt = 0; mt < 2; ++mt) {
#pragma unroll
      for (int v = 0; v < 8; ++v) {
        const float f = fmaxf(acc2[mt][t][v] + bias, 0.0f);
        const int e = base + mt * 16 + v + 8 * half;
        if (active && e < E) {
          edge_out[(size_t)e * 64 + 16 * t + laneM] = f;
          unsafeAtomicAdd(&agg[(size_t)ridx[mt][v] * 64 + 16 * t + laneM], f);
        }
      }
    }
  }
}

// --------------------------- node kernel -----------------------------------
// out = relu([x | agg] Wn1 + bn1) Wn2 + bn2 + x ; wave handles 32 nodes.

__global__ void __launch_bounds__(256)
node_kernel(const bf16x* __restrict__ xb, const bf16x* __restrict__ aggb,
            const float* __restrict__ xf,
            const bf16x* __restrict__ Wn1t, const float* __restrict__ bn1,
            const bf16x* __restrict__ Wn2t, const float* __restrict__ bn2,
            float* __restrict__ out, int Nn) {
  __shared__ bf16x sW1[128 * 64];
  __shared__ bf16x sW2[64 * 64];
  __shared__ bf16x sT[8 * 32 * LDS_STRIDE];

  stage_weights(sW1, Wn1t, 128 * 64, sW2, Wn2t, 64 * 64);

  const int wave  = threadIdx.x >> 5;
  const int lane  = threadIdx.x & 31;
  const int laneM = lane & 15;
  const int half  = lane >> 4;

  const int tiles32 = (Nn + 31) >> 5;
  int tile = blockIdx.x * 8 + wave;
  const bool active = (tile < tiles32);
  if (!active) tile = 0;
  const int base = tile * 32;

  int i0 = base + laneM;       if (i0 > Nn - 1) i0 = Nn - 1;
  int i1 = base + 16 + laneM;  if (i1 > Nn - 1) i1 = Nn - 1;
  const bf16x* xr0 = xb   + (size_t)i0 * 64;
  const bf16x* gr0 = aggb + (size_t)i0 * 64;
  const bf16x* xr1 = xb   + (size_t)i1 * 64;
  const bf16x* gr1 = aggb + (size_t)i1 * 64;

  // ---- layer 1 ----
  v8f acc[2][4] = {};
#pragma unroll
  for (int kt = 0; kt < 4; ++kt) {
    const int k = kt * 32;
    v16bf a0 = (k < 64) ? load_a16(xr0, k, half) : load_a16(gr0, k - 64, half);
    v16bf a1 = (k < 64) ? load_a16(xr1, k, half) : load_a16(gr1, k - 64, half);
#pragma unroll
    for (int t = 0; t < 4; ++t) {
      v16bf b = load_b16(sW1, 128, 16 * t + laneM, k, half);
      acc[0][t] = wmma_bf16(a0, b, acc[0][t]);
      acc[1][t] = wmma_bf16(a1, b, acc[1][t]);
    }
  }

  bf16x* myT = sT + wave * 32 * LDS_STRIDE;
#pragma unroll
  for (int t = 0; t < 4; ++t) {
    const float bias = bn1[16 * t + laneM];
#pragma unroll
    for (int mt = 0; mt < 2; ++mt) {
#pragma unroll
      for (int v = 0; v < 8; ++v) {
        float h = fmaxf(acc[mt][t][v] + bias, 0.0f);
        myT[(mt * 16 + v + 8 * half) * LDS_STRIDE + 16 * t + laneM] = (bf16x)h;
      }
    }
  }
  __syncthreads();

  // ---- layer 2 ----
  v8f acc2[2][4] = {};
  const bf16x* at0 = myT + laneM * LDS_STRIDE;
  const bf16x* at1 = myT + (16 + laneM) * LDS_STRIDE;
#pragma unroll
  for (int kt = 0; kt < 2; ++kt) {
    const int k = kt * 32;
    v16bf a0 = load_a16(at0, k, half);
    v16bf a1 = load_a16(at1, k, half);
#pragma unroll
    for (int t = 0; t < 4; ++t) {
      v16bf b = load_b16(sW2, 64, 16 * t + laneM, k, half);
      acc2[0][t] = wmma_bf16(a0, b, acc2[0][t]);
      acc2[1][t] = wmma_bf16(a1, b, acc2[1][t]);
    }
  }

#pragma unroll
  for (int t = 0; t < 4; ++t) {
    const float bias = bn2[16 * t + laneM];
#pragma unroll
    for (int mt = 0; mt < 2; ++mt) {
#pragma unroll
      for (int v = 0; v < 8; ++v) {
        const int i = base + mt * 16 + v + 8 * half;
        if (active && i < Nn) {
          const size_t off = (size_t)i * 64 + 16 * t + laneM;
          out[off] = acc2[mt][t][v] + bias + xf[off];   // fp32 residual
        }
      }
    }
  }
}

// --------------------------- launch ----------------------------------------

extern "C" void kernel_launch(void* const* d_in, const int* in_sizes, int n_in,
                              void* d_out, int out_size, void* d_ws, size_t ws_size,
                              hipStream_t stream) {
  const float* x   = (const float*)d_in[0];
  const int*   row = (const int*)d_in[1];
  const int*   col = (const int*)d_in[2];
  const float* We1 = (const float*)d_in[3];
  const float* be1 = (const float*)d_in[4];
  const float* We2 = (const float*)d_in[5];
  const float* be2 = (const float*)d_in[6];
  const float* Wn1 = (const float*)d_in[7];
  const float* bn1 = (const float*)d_in[8];
  const float* Wn2 = (const float*)d_in[9];
  const float* bn2 = (const float*)d_in[10];

  const int D = 64;
  const int N = in_sizes[0] / D;
  const int E = in_sizes[1];

  float* out_nodes = (float*)d_out;               // [N, 64]
  float* edge_feat = out_nodes + (size_t)N * D;   // [E, 64]

  char* p = (char*)d_ws;
  auto alloc = [&](size_t bytes) -> char* {
    char* r = p;
    p += (bytes + 255) & ~(size_t)255;
    return r;
  };
  float* agg  = (float*)alloc((size_t)N * D * sizeof(float));
  bf16x* xbuf = (bf16x*)alloc((size_t)N * D * sizeof(bf16x));
  bf16x* aggb = (bf16x*)alloc((size_t)N * D * sizeof(bf16x));
  bf16x* We1t = (bf16x*)alloc((size_t)128 * 64 * sizeof(bf16x));
  bf16x* We2t = (bf16x*)alloc((size_t)64 * 64 * sizeof(bf16x));
  bf16x* Wn1t = (bf16x*)alloc((size_t)128 * 64 * sizeof(bf16x));
  bf16x* Wn2t = (bf16x*)alloc((size_t)64 * 64 * sizeof(bf16x));

  {
    int n = N * D;
    prep_x_agg<<<(n + 255) / 256, 256, 0, stream>>>(x, xbuf, agg, n);
  }
  prep_w<<<(128 * 64 + 255) / 256, 256, 0, stream>>>(We1, We1t, 128, 64);
  prep_w<<<(64 * 64 + 255) / 256, 256, 0, stream>>>(We2, We2t, 64, 64);
  prep_w<<<(128 * 64 + 255) / 256, 256, 0, stream>>>(Wn1, Wn1t, 128, 64);
  prep_w<<<(64 * 64 + 255) / 256, 256, 0, stream>>>(Wn2, Wn2t, 64, 64);

  {
    int tiles  = (E + 31) / 32;
    int blocks = (tiles + 7) / 8;   // 8 waves x 32 edges per 256-thread block
    edge_kernel<<<blocks, 256, 0, stream>>>(xbuf, row, col, We1t, be1, We2t, be2,
                                            edge_feat, agg, E);
  }
  {
    int n = N * D;
    cvt_f2b<<<(n + 255) / 256, 256, 0, stream>>>(agg, aggb, n);
  }
  {
    int tiles  = (N + 31) / 32;
    int blocks = (tiles + 7) / 8;
    node_kernel<<<blocks, 256, 0, stream>>>(xbuf, aggb, x, Wn1t, bn1, Wn2t, bn2,
                                            out_nodes, N);
  }
}
